// CausalSelfAttention_87960930222535
// MI455X (gfx1250) — matrix operand order repaired and checked by reading, not compiled, not run on hardware
//
#include <hip/hip_runtime.h>

// ---------------------------------------------------------------------------
// GPT-2 causal self-attention block for MI455X (gfx1250, wave32, WMMA).
// All matmuls run through V_WMMA_F32_16X16X32_BF16 (f32 accumulate).
//   k0a: Xb  = bf16(x)                       (pack once, stream many)
//   k0b: WTa = bf16(W_attn^T), WTp = bf16(W_proj^T)
//   k1:  qkv = Xb @ WTa^T + b  -> bf16 Q/K/V in [B,H,S,64]
//   k2:  flash attention (S^T = K Q^T trick -> in-lane softmax) -> bf16 [B,S,768]
//   k3:  y = O @ WTp^T + b -> fp32
// GEMMs: 256 thr, block tile 64x256, wave tile 32x64 (2x4 WMMA subtiles),
// BK=64, double-buffered LDS (80 KB), async global->LDS staging (ASYNCcnt),
// last iteration peeled so the steady-state loop has no branches.
// ---------------------------------------------------------------------------

typedef __bf16 bf16;
typedef __attribute__((ext_vector_type(2)))  bf16  bf16x2;
typedef __attribute__((ext_vector_type(4)))  bf16  bf16x4;
typedef __attribute__((ext_vector_type(8)))  bf16  bf16x8;
typedef __attribute__((ext_vector_type(16))) bf16  bf16x16;
typedef __attribute__((ext_vector_type(2)))  float f32x2;
typedef __attribute__((ext_vector_type(4)))  float f32x4;
typedef __attribute__((ext_vector_type(8)))  float f32x8;
typedef __attribute__((ext_vector_type(8)))  int   i32x8;

#define WMMA_BF16(A, B, C) \
    __builtin_amdgcn_wmma_f32_16x16x32_bf16(false, (A), false, (B), (short)0, (C), false, false)

// 32-bit LDS offset of a __shared__ object (AS3 pointers are LDS byte addresses).
static __device__ inline unsigned lds_addr(const void* p) {
    return (unsigned)(unsigned long long)
        (__attribute__((address_space(3))) const void*)p;
}

// Async global->LDS 16-byte copy (per lane), tracked by ASYNCcnt.
#define ASYNC_LDS_B128(ldsoff, gaddr)                              \
    asm volatile("global_load_async_to_lds_b128 %0, %1, off"       \
                 :: "v"(ldsoff), "v"(gaddr) : "memory")

#define WAIT_ASYNC0() asm volatile("s_wait_asynccnt 0" ::: "memory")

static __device__ inline f32x8 zero8() {
    f32x8 z;
#pragma unroll
    for (int i = 0; i < 8; ++i) z[i] = 0.0f;
    return z;
}

static __device__ inline bf16x16 join16(bf16x8 lo, bf16x8 hi) {
    bf16x16 r;
#pragma unroll
    for (int j = 0; j < 8; ++j) { r[j] = lo[j]; r[j + 8] = hi[j]; }
    return r;
}

// A-operand (16x32 bf16): lane half 0 -> K 0..15 ; half 1 -> K 16..31.
static __device__ inline bf16x16 load_opA(const bf16* row, int half, int koff) {
    bf16x8 lo = *(const bf16x8*)(row + koff + half * 16);
    bf16x8 hi = *(const bf16x8*)(row + koff + 8 + half * 16);
    return join16(lo, hi);
}

// B-operand (32x16 bf16): lane half 0 -> K 0..15 ; half 1 -> K 16..31 of column n.
static __device__ inline bf16x16 load_opB(const bf16* col, int half, int koff) {
    const bf16* p = col + koff + 16 * half;
    return join16(*(const bf16x8*)p, *(const bf16x8*)(p + 8));
}

// ===========================================================================
// Pre-pass A: X fp32 -> bf16 streaming convert (16 elems / thread).
// ===========================================================================
__global__ __launch_bounds__(256)
void convert_x_kernel(const float* __restrict__ X, bf16* __restrict__ Xb)
{
    size_t base = ((size_t)blockIdx.x * 256 + threadIdx.x) * 16;
#pragma unroll
    for (int j = 0; j < 2; ++j) {
        f32x4 a = *(const f32x4*)(X + base + j * 8);
        f32x4 b = *(const f32x4*)(X + base + j * 8 + 4);
        bf16x8 r;
        bf16x4 ra = __builtin_convertvector(a, bf16x4);
        bf16x4 rb = __builtin_convertvector(b, bf16x4);
#pragma unroll
        for (int i = 0; i < 4; ++i) { r[i] = ra[i]; r[i + 4] = rb[i]; }
        *(bf16x8*)(Xb + base + j * 8) = r;
    }
}

// ===========================================================================
// Pre-pass B: W[K x N] fp32 -> WT[N x K] bf16, LDS-tiled 64x64 transpose.
// grid = (N/64, K/64).  K == 768 for both weights here.
// ===========================================================================
__global__ __launch_bounds__(256)
void transpose_w_kernel(const float* __restrict__ W, bf16* __restrict__ WT, int N)
{
    __shared__ bf16 sh[64 * 72];           // [k][n], padded rows
    const int tid = threadIdx.x;
    const int n0  = blockIdx.x * 64;
    const int k0  = blockIdx.y * 64;

#pragma unroll
    for (int r = 0; r < 4; ++r) {          // 64x64 fp32 in, bf16 to LDS
        int idx = tid + 256 * r;
        int kk = idx >> 4, n4 = (idx & 15) * 4;
        f32x4 w = *(const f32x4*)(W + (size_t)(k0 + kk) * N + n0 + n4);
        *(bf16x4*)(&sh[kk * 72 + n4]) = __builtin_convertvector(w, bf16x4);
    }
    __syncthreads();
#pragma unroll
    for (int j = 0; j < 2; ++j) {          // transposed bf16x8 out
        int idx = tid * 2 + j;
        int nn = idx >> 3, c8 = (idx & 7) * 8;
        bf16x8 v;
#pragma unroll
        for (int i = 0; i < 8; ++i) v[i] = sh[(c8 + i) * 72 + nn];
        *(bf16x8*)(WT + (size_t)(n0 + nn) * 768 + k0 + c8) = v;
    }
}

// ===========================================================================
// Kernel 1: QKV projection.  C[8192, 2304] = Xb[8192,768] @ WTa^T + b
// ===========================================================================
__global__ __launch_bounds__(256)
void qkv_gemm_kernel(const bf16* __restrict__ Xb, const bf16* __restrict__ WT,
                     const float* __restrict__ bias,
                     bf16* __restrict__ Qb, bf16* __restrict__ Kb, bf16* __restrict__ Vb)
{
    __shared__ bf16 As[2][64 * 64];    // [m][k]
    __shared__ bf16 Bs[2][256 * 64];   // [n][k]

    const int tid  = threadIdx.x;
    const int lane = tid & 31;
    const int wid  = tid >> 5;
    const int half = lane >> 4;
    const int lq   = lane & 15;
    const int m0   = blockIdx.y * 64;
    const int n0   = blockIdx.x * 256;
    const int wm   = wid >> 2;      // 0..1 -> 32 rows
    const int wn   = wid & 3;       // 0..3 -> 64 cols

    f32x8 acc[2][4];
#pragma unroll
    for (int a = 0; a < 2; ++a)
#pragma unroll
        for (int c = 0; c < 4; ++c) acc[a][c] = zero8();

    auto stage = [&](int k0, int buf) {
#pragma unroll
        for (int j = 0; j < 2; ++j) {       // A tile: 64x64, async 16B copies
            int idx = tid * 2 + j;
            int row = idx >> 3, c8 = (idx & 7) * 8;
            ASYNC_LDS_B128(lds_addr(&As[buf][row * 64 + c8]),
                           Xb + (size_t)(m0 + row) * 768 + k0 + c8);
        }
#pragma unroll
        for (int r = 0; r < 8; ++r) {       // B tile: 256x64, async 16B copies
            int idx = tid + 256 * r;
            int nr = idx >> 3, c8 = (idx & 7) * 8;
            ASYNC_LDS_B128(lds_addr(&Bs[buf][nr * 64 + c8]),
                           WT + (size_t)(n0 + nr) * 768 + k0 + c8);
        }
    };

    auto compute = [&](int kt) {
        const int buf = kt & 1;
#pragma unroll
        for (int kk = 0; kk < 64; kk += 32) {
            bf16x16 a0 = load_opA(&As[buf][(wm * 32 + lq) * 64], half, kk);
            bf16x16 a1 = load_opA(&As[buf][(wm * 32 + 16 + lq) * 64], half, kk);
#pragma unroll
            for (int ni = 0; ni < 4; ++ni) {
                bf16x16 b = load_opB(&Bs[buf][(wn * 64 + ni * 16 + lq) * 64], half, kk);
                acc[0][ni] = WMMA_BF16(a0, b, acc[0][ni]);
                acc[1][ni] = WMMA_BF16(a1, b, acc[1][ni]);
            }
        }
    };

    stage(0, 0);
    for (int kt = 0; kt < 11; ++kt) {       // steady state: branch-free body
        WAIT_ASYNC0();
        __syncthreads();
        stage((kt + 1) * 64, (kt + 1) & 1);
        compute(kt);
    }
    WAIT_ASYNC0();                          // peeled last stage
    __syncthreads();
    compute(11);

    // ---- epilogue: +bias, scatter into head-major Q/K/V [B,H,S,64] bf16 ----
#pragma unroll
    for (int ni = 0; ni < 4; ++ni) {
        int f = n0 + wn * 64 + ni * 16 + lq;      // output feature in [0,2304)
        float bv = bias[f];
        int which = f / 768, e = f % 768;
        int h = e >> 6, dd = e & 63;
        bf16* dst = (which == 0) ? Qb : (which == 1) ? Kb : Vb;
#pragma unroll
        for (int mi = 0; mi < 2; ++mi) {
            int m_base = m0 + wm * 32 + mi * 16 + 8 * half;  // first token of this strip
            int bb = m_base >> 12, s = m_base & 4095;        // strip stays in one batch
            size_t base = (size_t)((bb * 12 + h) * 4096 + s) * 64 + dd;
#pragma unroll
            for (int i = 0; i < 8; ++i)
                dst[base + (size_t)i * 64] = (bf16)(acc[mi][ni][i] + bv);
        }
    }
}

// ===========================================================================
// Kernel 2: flash attention.  Block = 128 thr (4 waves); wave owns 16 queries.
// S^T = K Q^T -> in-lane softmax; O^T = V^T P; double-buffered K/V tiles.
// ===========================================================================
__global__ __launch_bounds__(128)
void attn_kernel(const bf16* __restrict__ Qb, const bf16* __restrict__ Kb,
                 const bf16* __restrict__ Vb, bf16* __restrict__ Ob)
{
    __shared__ bf16 Ks[2][32 * 64];   // [key][d]
    __shared__ bf16 Vt[2][64 * 32];   // [d][key]  (V transposed)

    const int tid  = threadIdx.x;
    const int lane = tid & 31;
    const int wid  = tid >> 5;
    const int half = lane >> 4;
    const int lq   = lane & 15;
    const int h  = blockIdx.y;
    const int b  = blockIdx.z;
    const int bh = b * 12 + h;
    const int q0 = blockIdx.x * 64 + wid * 16;
    const int q  = q0 + lq;                      // this lane's query column

    const bf16* Qp = Qb + (size_t)bh * 4096 * 64;
    const bf16* Kp = Kb + (size_t)bh * 4096 * 64;
    const bf16* Vp = Vb + (size_t)bh * 4096 * 64;

    // Q^T is the B operand of the scores WMMA: lane n holds column n = Q row q.
    bf16x16 Bq0, Bq1;
    {
        const bf16* r = Qp + (size_t)q * 64;
        Bq0 = load_opB(r, half, 0);
        Bq1 = load_opB(r, half, 32);
    }

    float m_old = -1e30f, l = 0.0f;
    f32x8 accO[4];
#pragma unroll
    for (int dt = 0; dt < 4; ++dt) accO[dt] = zero8();

    auto stageKV = [&](int kb, int buf) {
        const int kstart = kb * 32;
        {   // K tile: async 16B copies (already bf16, row-major)
            int off = tid * 16;
            ASYNC_LDS_B128(lds_addr(&Ks[buf][off]),
                           Kp + (size_t)kstart * 64 + off);
            ASYNC_LDS_B128(lds_addr(&Ks[buf][off + 8]),
                           Kp + (size_t)kstart * 64 + off + 8);
        }
        {   // V transpose: 2 adjacent keys x 8 dims per thread -> bf16x2 stores
            int kt = (tid & 15) * 2;
            int d0 = (tid >> 4) * 8;
            bf16x8 va = *(const bf16x8*)(Vp + (size_t)(kstart + kt) * 64 + d0);
            bf16x8 vb = *(const bf16x8*)(Vp + (size_t)(kstart + kt + 1) * 64 + d0);
#pragma unroll
            for (int j = 0; j < 8; ++j) {
                bf16x2 p = { va[j], vb[j] };
                *(bf16x2*)(&Vt[buf][(d0 + j) * 32 + kt]) = p;
            }
        }
    };

    auto compute = [&](int kb) {
        const int kstart = kb * 32;
        const int buf = kb & 1;
        if (kstart <= q0 + 15) {                 // wave-uniform; EXEC stays all-ones
            // ---- scores S^T: two 16(key)x16(query) tiles, K-dim = d = 64 ----
            f32x8 s0 = zero8(), s1 = zero8();
            const bf16* K0 = &Ks[buf][lq * 64];
            const bf16* K1 = &Ks[buf][(16 + lq) * 64];
            s0 = WMMA_BF16(load_opA(K0, half, 0),  Bq0, s0);
            s0 = WMMA_BF16(load_opA(K0, half, 32), Bq1, s0);
            s1 = WMMA_BF16(load_opA(K1, half, 0),  Bq0, s1);
            s1 = WMMA_BF16(load_opA(K1, half, 32), Bq1, s1);

            // ---- causal mask + online softmax (lane owns column q) ----
            float p0[8], p1[8];
            float mb = -1e30f;
#pragma unroll
            for (int i = 0; i < 8; ++i) {
                int k0i = kstart + i + 8 * half;     // tile0 key
                int k1i = k0i + 16;                  // tile1 key
                float v0 = s0[i] * 0.125f; if (k0i > q) v0 = -1e30f;
                float v1 = s1[i] * 0.125f; if (k1i > q) v1 = -1e30f;
                p0[i] = v0; p1[i] = v1;
                mb = fmaxf(mb, fmaxf(v0, v1));
            }
            mb = fmaxf(mb, __shfl_xor(mb, 16, 32));  // other half of this column
            float m_new = fmaxf(m_old, mb);
            float alpha = __expf(m_old - m_new);
            float rs = 0.0f;
#pragma unroll
            for (int i = 0; i < 8; ++i) {
                p0[i] = __expf(p0[i] - m_new);
                p1[i] = __expf(p1[i] - m_new);
                rs += p0[i] + p1[i];
            }
            rs += __shfl_xor(rs, 16, 32);
            l = l * alpha + rs;
            m_old = m_new;
#pragma unroll
            for (int dt = 0; dt < 4; ++dt)
#pragma unroll
                for (int i = 0; i < 8; ++i) accO[dt][i] *= alpha;

            // ---- repack P (D layout) into B-operand layout via lane exchange ----
            int pk0[4], pk1[4];
#pragma unroll
            for (int j = 0; j < 4; ++j) {
                f32x2 a0 = { p0[2 * j], p0[2 * j + 1] };
                f32x2 a1 = { p1[2 * j], p1[2 * j + 1] };
                pk0[j] = __builtin_bit_cast(int, __builtin_convertvector(a0, bf16x2));
                pk1[j] = __builtin_bit_cast(int, __builtin_convertvector(a1, bf16x2));
            }
            i32x8 bu;
#pragma unroll
            for (int j = 0; j < 4; ++j) {
                int o0 = __shfl_xor(pk0[j], 16, 32);
                int o1 = __shfl_xor(pk1[j], 16, 32);
                bu[j]     = half ? o1     : pk0[j];
                bu[4 + j] = half ? pk1[j] : o0;
            }
            bf16x16 Bp = __builtin_bit_cast(bf16x16, bu);

            // ---- O^T += V^T @ P : 4 d-tiles, K-dim = 32 keys ----
#pragma unroll
            for (int dt = 0; dt < 4; ++dt) {
                const bf16* vr = &Vt[buf][(dt * 16 + lq) * 32];
                accO[dt] = WMMA_BF16(load_opA(vr, half, 0), Bp, accO[dt]);
            }
        }
    };

    const int nkb = blockIdx.x * 2 + 2;          // key blocks of 32 up to diagonal
    stageKV(0, 0);
    for (int kb = 0; kb < nkb - 1; ++kb) {       // steady state: branch-free staging
        WAIT_ASYNC0();
        __syncthreads();
        stageKV(kb + 1, (kb + 1) & 1);
        compute(kb);
    }
    WAIT_ASYNC0();                               // peeled last block
    __syncthreads();
    compute(nkb - 1);

    // ---- finalize: O = O^T / l, store bf16 into [B,S,768] ----
    float inv = 1.0f / l;
    bf16* orow = Ob + (size_t)(b * 4096 + q) * 768 + h * 64;
#pragma unroll
    for (int dt = 0; dt < 4; ++dt)
#pragma unroll
        for (int i = 0; i < 8; ++i)
            orow[dt * 16 + i + 8 * half] = (bf16)(accO[dt][i] * inv);
}

// ===========================================================================
// Kernel 3: output projection.  y[8192,768] = O[8192,768] @ WTp^T + b (fp32 out)
// ===========================================================================
__global__ __launch_bounds__(256)
void proj_gemm_kernel(const bf16* __restrict__ O, const bf16* __restrict__ WT,
                      const float* __restrict__ bias, float* __restrict__ out)
{
    __shared__ bf16 As[2][64 * 64];
    __shared__ bf16 Bs[2][256 * 64];

    const int tid  = threadIdx.x;
    const int lane = tid & 31;
    const int wid  = tid >> 5;
    const int half = lane >> 4;
    const int lq   = lane & 15;
    const int m0   = blockIdx.y * 64;
    const int n0   = blockIdx.x * 256;
    const int wm   = wid >> 2;
    const int wn   = wid & 3;

    f32x8 acc[2][4];
#pragma unroll
    for (int a = 0; a < 2; ++a)
#pragma unroll
        for (int c = 0; c < 4; ++c) acc[a][c] = zero8();

    auto stage = [&](int k0, int buf) {
#pragma unroll
        for (int j = 0; j < 2; ++j) {
            int idx = tid * 2 + j;
            int row = idx >> 3, c8 = (idx & 7) * 8;
            ASYNC_LDS_B128(lds_addr(&As[buf][row * 64 + c8]),
                           O + (size_t)(m0 + row) * 768 + k0 + c8);
        }
#pragma unroll
        for (int r = 0; r < 8; ++r) {
            int idx = tid + 256 * r;
            int nr = idx >> 3, c8 = (idx & 7) * 8;
            ASYNC_LDS_B128(lds_addr(&Bs[buf][nr * 64 + c8]),
                           WT + (size_t)(n0 + nr) * 768 + k0 + c8);
        }
    };

    auto compute = [&](int kt) {
        const int buf = kt & 1;
#pragma unroll
        for (int kk = 0; kk < 64; kk += 32) {
            bf16x16 a0 = load_opA(&As[buf][(wm * 32 + lq) * 64], half, kk);
            bf16x16 a1 = load_opA(&As[buf][(wm * 32 + 16 + lq) * 64], half, kk);
#pragma unroll
            for (int ni = 0; ni < 4; ++ni) {
                bf16x16 b = load_opB(&Bs[buf][(wn * 64 + ni * 16 + lq) * 64], half, kk);
                acc[0][ni] = WMMA_BF16(a0, b, acc[0][ni]);
                acc[1][ni] = WMMA_BF16(a1, b, acc[1][ni]);
            }
        }
    };

    stage(0, 0);
    for (int kt = 0; kt < 11; ++kt) {
        WAIT_ASYNC0();
        __syncthreads();
        stage((kt + 1) * 64, (kt + 1) & 1);
        compute(kt);
    }
    WAIT_ASYNC0();
    __syncthreads();
    compute(11);

#pragma unroll
    for (int ni = 0; ni < 4; ++ni) {
        int f = n0 + wn * 64 + ni * 16 + lq;
        float bv = bias[f];
#pragma unroll
        for (int mi = 0; mi < 2; ++mi) {
            int m_base = m0 + wm * 32 + mi * 16 + 8 * half;
            size_t base = (size_t)m_base * 768 + f;
#pragma unroll
            for (int i = 0; i < 8; ++i)
                out[base + (size_t)i * 768] = acc[mi][ni][i] + bv;
        }
    }
}

// ===========================================================================
extern "C" void kernel_launch(void* const* d_in, const int* in_sizes, int n_in,
                              void* d_out, int out_size, void* d_ws, size_t ws_size,
                              hipStream_t stream) {
    const float* x      = (const float*)d_in[0];   // [2,4096,768]
    const float* W_attn = (const float*)d_in[1];   // [768,2304]
    const float* b_attn = (const float*)d_in[2];   // [2304]
    const float* W_proj = (const float*)d_in[3];   // [768,768]
    const float* b_proj = (const float*)d_in[4];   // [768]
    float* out = (float*)d_out;                    // [2,4096,768] fp32

    char* ws = (char*)d_ws;
    const size_t qkvBytes = (size_t)2 * 12 * 4096 * 64 * sizeof(bf16); // 12.58 MB
    bf16* Qb  = (bf16*)(ws);
    bf16* Kb  = (bf16*)(ws + qkvBytes);
    bf16* Vb  = (bf16*)(ws + 2 * qkvBytes);
    bf16* Ob  = (bf16*)(ws + 3 * qkvBytes);
    bf16* Xb  = (bf16*)(ws + 4 * qkvBytes);        // [8192,768] bf16 (same size)
    bf16* WTa = (bf16*)(ws + 5 * qkvBytes);        // [2304,768] bf16 (3.54 MB)
    bf16* WTp = (bf16*)(ws + 5 * qkvBytes + (size_t)2304 * 768 * sizeof(bf16));

    // 0) pack inputs once: X -> bf16, W -> bf16 transposed
    convert_x_kernel<<<dim3(1536), 256, 0, stream>>>(x, Xb);
    transpose_w_kernel<<<dim3(36, 12), 256, 0, stream>>>(W_attn, WTa, 2304);
    transpose_w_kernel<<<dim3(12, 12), 256, 0, stream>>>(W_proj, WTp, 768);
    // 1) QKV projection: M=8192, N=2304 -> grid (2304/256, 8192/64)
    qkv_gemm_kernel<<<dim3(9, 128), 256, 0, stream>>>(Xb, WTa, b_attn, Qb, Kb, Vb);
    // 2) flash attention: (S/64, H, B)
    attn_kernel<<<dim3(64, 12, 2), 128, 0, stream>>>(Qb, Kb, Vb, Ob);
    // 3) output projection: M=8192, N=768 -> grid (768/256, 8192/64)
    proj_gemm_kernel<<<dim3(3, 128), 256, 0, stream>>>(Ob, WTp, b_proj, out);
}
